// ConvQCL_2499670966554
// MI455X (gfx1250) — matrix operand order repaired and checked by reading, not compiled, not run on hardware
//
#include <hip/hip_runtime.h>

// ---------------- WMMA types / helpers (CDNA5 gfx1250, wave32) ----------------
typedef _Float16 h16 __attribute__((ext_vector_type(16)));
typedef _Float16 h8v __attribute__((ext_vector_type(8)));
typedef float    f8  __attribute__((ext_vector_type(8)));

// A-fragment (16x32 f16): per ISA, lane holds row m=lane%16; lanes h=lane/16 get
// K = {8h+e, 16+8h+e}.  Caller passes p = row_base + k0 + 8h; chunks at +0 / +16.
__device__ inline void load_fragA(const _Float16* __restrict__ p, h16& out) {
  h8v a = *(const h8v*)p;
  h8v b = *(const h8v*)(p + 16);
#pragma unroll
  for (int e = 0; e < 8; ++e) { out[e] = a[e]; out[8 + e] = b[e]; }
}

// B-fragment (32x16 f16): lane n=lane%16 is the column; K = {8h+e, 16+8h+e}.
// Caller passes p = col_base + k0 + 8h; two h8 loads at +0 / +16.
__device__ inline void load_fragB(const _Float16* __restrict__ p, h16& out) {
  h8v a = *(const h8v*)p;
  h8v b = *(const h8v*)(p + 16);
#pragma unroll
  for (int e = 0; e < 8; ++e) { out[e] = a[e]; out[8 + e] = b[e]; }
}

// Store C/D fragment (16x16 f32) to row-major [.,256].
__device__ inline void store_d(float* __restrict__ D, int tm, int tn, int lane,
                               const f8& c) {
  int hh = (lane >> 4) & 1;
  int n  = tn * 16 + (lane & 15);
#pragma unroll
  for (int j = 0; j < 8; ++j) D[(size_t)(tm * 16 + j + 8 * hh) * 256 + n] = c[j];
}

// f16x3 error-compensated product: C += Ahi*Bhi + Ahi*Blo + Alo*Bhi (f32 acc).
__device__ inline f8 wmma3(const h16& ah, const h16& al, const h16& bh,
                           const h16& bl, f8 c) {
  c = __builtin_amdgcn_wmma_f32_16x16x32_f16(false, ah, false, bh, (short)0, c, false, false);
  c = __builtin_amdgcn_wmma_f32_16x16x32_f16(false, ah, false, bl, (short)0, c, false, false);
  c = __builtin_amdgcn_wmma_f32_16x16x32_f16(false, al, false, bh, (short)0, c, false, false);
  return c;
}

__device__ inline void split_store(_Float16* __restrict__ hiP,
                                   _Float16* __restrict__ loP, int o, float v) {
  _Float16 h = (_Float16)v;
  hiP[o] = h;
  loP[o] = (_Float16)(v - (float)h);
}

// ---------------- Kernel 1: conv cells (9-qubit, all-real, LDS-resident) ------
// One workgroup (256 threads) per patch; state = 512 floats in LDS.
// CNOT chain == gather from idx ^ (idx>>1).  Output <Z_0> -> cv[p].
__global__ __launch_bounds__(256) void qconv_kernel(const float* __restrict__ x,
                                                    const float* __restrict__ wc,
                                                    int cdepth, float* __restrict__ cv) {
  int p = blockIdx.x, t = threadIdx.x;
  int b = p >> 6, yy = (p >> 3) & 7, xx = p & 7;
  __shared__ float st[512];
  __shared__ float cs[9], sn[9];
  __shared__ float red[256];

  if (t < 9) {
    int dy = t / 3, dx = t % 3;
    int iy = yy + dy - 1, ix = xx + dx - 1;
    float th = 0.f;
    if (iy >= 0 && iy < 8 && ix >= 0 && ix < 8)
      th = asinf(2.f * x[(size_t)b * 64 + iy * 8 + ix] - 1.f);
    cs[t] = cosf(0.5f * th); sn[t] = sinf(0.5f * th);
  }
  __syncthreads();

  // rank-1 product state |psi(theta)>
  for (int f = t; f < 512; f += 256) {
    float v = 1.f;
#pragma unroll
    for (int w = 0; w < 9; ++w) v *= ((f >> (8 - w)) & 1) ? sn[w] : cs[w];
    st[f] = v;
  }
  __syncthreads();

  for (int k = 0; k < cdepth; ++k) {
    // CNOT chain permutation: new[c] = old[c ^ (c>>1)]
    int f1 = t + 256;
    float t0 = st[t ^ (t >> 1)];
    float t1 = st[f1 ^ (f1 >> 1)];
    __syncthreads();
    st[t] = t0; st[f1] = t1;
    __syncthreads();
    // RY layer (real butterflies)
    for (int i = 0; i < 9; ++i) {
      float th = wc[k * 9 + i];
      float c = cosf(0.5f * th), s = sinf(0.5f * th);
      int m = 1 << (8 - i);
      float a0 = st[t],       b0 = st[t ^ m];
      float a1 = st[t + 256], b1 = st[(t + 256) ^ m];
      float n0 = (t & m)         ? (s * b0 + c * a0) : (c * a0 - s * b0);
      float n1 = ((t + 256) & m) ? (s * b1 + c * a1) : (c * a1 - s * b1);
      __syncthreads();
      st[t] = n0; st[t + 256] = n1;
      __syncthreads();
    }
  }
  // <Z_0>: + for MSB==0 (f<256), - otherwise
  red[t] = st[t] * st[t] - st[t + 256] * st[t + 256];
  __syncthreads();
  for (int off = 128; off > 0; off >>= 1) {
    if (t < off) red[t] += red[t + off];
    __syncthreads();
  }
  if (t == 0) cv[p] = red[0];
}

// ---------------- Kernel 2: 2x2 max pool -------------------------------------
__global__ void pool_kernel(const float* __restrict__ cv, float* __restrict__ pooled, int B) {
  int i = blockIdx.x * blockDim.x + threadIdx.x;
  if (i >= B * 16) return;
  int b = i >> 4, py = (i >> 2) & 3, px = i & 3;
  const float* c0 = cv + (size_t)b * 64;
  float m = c0[(py * 2) * 8 + px * 2];
  m = fmaxf(m, c0[(py * 2) * 8 + px * 2 + 1]);
  m = fmaxf(m, c0[(py * 2 + 1) * 8 + px * 2]);
  m = fmaxf(m, c0[(py * 2 + 1) * 8 + px * 2 + 1]);
  pooled[i] = m;
}

// ---------------- Kernel 3: build A_k / B_k = Kronecker RY matrices -----------
// mats plane layout: [(k*2+isB)][hi|lo][256*256] as f16; element = prod of 8 G factors.
__global__ __launch_bounds__(256) void build_mats_kernel(const float* __restrict__ wd,
                                                         _Float16* __restrict__ mats) {
  int matid = blockIdx.y;           // k*2 + isB
  int k = matid >> 1, isB = matid & 1;
  __shared__ float cs[8], sn[8];
  int t = threadIdx.x;
  if (t < 8) {
    float th = wd[k * 16 + isB * 8 + t];
    cs[t] = cosf(0.5f * th); sn[t] = sinf(0.5f * th);
  }
  __syncthreads();
  int idx = blockIdx.x * 256 + t;
  int ro = idx >> 8, ri = idx & 255;
  float v = 1.f;
#pragma unroll
  for (int i = 0; i < 8; ++i) {
    int ob = (ro >> (7 - i)) & 1, ib = (ri >> (7 - i)) & 1;
    v *= (ob == ib) ? cs[i] : (ib ? -sn[i] : sn[i]);
  }
  _Float16* base = mats + (size_t)matid * 2 * 65536;
  split_store(base, base + 65536, idx, v);
}

// ---------------- Kernel 4: dense cell (16-qubit) fully fused, WMMA GEMMs -----
// One workgroup (256 thr = 8 waves) per sample. Per layer:
//   split pass 1: St[n][k] = f16hi/lo( S[gray((k<<8)|n)] )   (perm+transpose+split)
//   GEMM1:        T = A_k x St            (pure f16 loads + WMMA)
//   split pass 2: Tp[o]    = f16hi/lo( T[o] )                 (row-major split)
//   GEMM2:        S = Tp x B_k^T
// each GEMM is 256x256x256 x {re,im} via f16x3 WMMA. Then marginals + linear head.
__global__ __launch_bounds__(256) void qdense_kernel(
    const float* __restrict__ pooled, const _Float16* __restrict__ mats,
    float* __restrict__ SreB, float* __restrict__ SimB,
    float* __restrict__ TreB, float* __restrict__ TimB,
    _Float16* __restrict__ SpB, _Float16* __restrict__ TpB,
    const float* __restrict__ w_out, const float* __restrict__ b_out,
    float* __restrict__ out, int ddepth, int out_dim) {
  int s = blockIdx.x, t = threadIdx.x;
  int lane = t & 31, wave = t >> 5;
  float* sre = SreB + (size_t)s * 65536;
  float* sim = SimB + (size_t)s * 65536;
  float* tre = TreB + (size_t)s * 65536;
  float* tim = TimB + (size_t)s * 65536;
  // split-plane order: reHi, reLo, imHi, imLo
  _Float16* spReHi = SpB + (size_t)s * 4 * 65536;
  _Float16* spReLo = spReHi + 65536;
  _Float16* spImHi = spReHi + 2 * 65536;
  _Float16* spImLo = spReHi + 3 * 65536;
  _Float16* tpReHi = TpB + (size_t)s * 4 * 65536;
  _Float16* tpReLo = tpReHi + 65536;
  _Float16* tpImHi = tpReHi + 2 * 65536;
  _Float16* tpImLo = tpReHi + 3 * 65536;

  __shared__ float q0r[16], q0i[16], q1r[16], q1i[16];
  __shared__ float fw[16];
  __shared__ float red2[16 * 256];

  if (t < 16) {
    float v = fminf(1.f, fmaxf(-1.f, pooled[s * 16 + t]));
    float a = asinf(v), ph = acosf(v * v);
    float ca = cosf(0.5f * a),  sa = sinf(0.5f * a);
    float cp = cosf(0.5f * ph), sp = sinf(0.5f * ph);
    q0r[t] = ca * cp; q0i[t] = -ca * sp;   // e^{-i phi/2} cos(a/2)
    q1r[t] = sa * cp; q1i[t] =  sa * sp;   // e^{+i phi/2} sin(a/2)
  }
  __syncthreads();

  // initial product state (complex)
  for (int f = t; f < 65536; f += 256) {
    float re = 1.f, im = 0.f;
#pragma unroll
    for (int w = 0; w < 16; ++w) {
      int bb = (f >> (15 - w)) & 1;
      float r = bb ? q1r[w] : q0r[w];
      float i = bb ? q1i[w] : q0i[w];
      float nr = re * r - im * i;
      im = re * i + im * r;
      re = nr;
    }
    sre[f] = re; sim[f] = im;
  }
  __threadfence();
  __syncthreads();

  for (int k = 0; k < ddepth; ++k) {
    const _Float16* Ah = mats + (size_t)(k * 2 + 0) * 2 * 65536;
    const _Float16* Al = Ah + 65536;
    const _Float16* Bh = mats + (size_t)(k * 2 + 1) * 2 * 65536;
    const _Float16* Bl = Bh + 65536;

    // ---- split pass 1: gray-permute + transpose + hi/lo split of S ----
    // target o = n*256 + kk  (coalesced writes; gathered f32 reads hit L2)
    for (int o = t; o < 65536; o += 256) {
      int n = o >> 8, kk = o & 255;
      int idx = (kk << 8) | n;
      int src = idx ^ (idx >> 1);            // CNOT-chain permutation
      split_store(spReHi, spReLo, o, sre[src]);
      split_store(spImHi, spImLo, o, sim[src]);
    }
    __threadfence();
    __syncthreads();

    // ---- GEMM1: T = A * gray(S)  (all-f16 contiguous fragment loads) ----
    for (int tile = wave; tile < 256; tile += 8) {
      int tm = tile >> 4, tn = tile & 15;
      int hh = (lane >> 4) & 1, lan = lane & 15;
      const _Float16* aBase = Ah + (size_t)(tm * 16 + lan) * 256;
      const _Float16* aBaseL = Al + (size_t)(tm * 16 + lan) * 256;
      size_t bOff = (size_t)(tn * 16 + lan) * 256 + 8 * hh;    // St is [n][k]
      __builtin_prefetch(aBase, 0, 0);
      f8 cr = {}, ci = {};
      for (int k0 = 0; k0 < 256; k0 += 32) {
        h16 ah, al, brh, brl, bih, bil;
        load_fragA(aBase + k0 + 8 * hh, ah);
        load_fragA(aBaseL + k0 + 8 * hh, al);
        load_fragB(spReHi + bOff + k0, brh);
        load_fragB(spReLo + bOff + k0, brl);
        load_fragB(spImHi + bOff + k0, bih);
        load_fragB(spImLo + bOff + k0, bil);
        cr = wmma3(ah, al, brh, brl, cr);
        ci = wmma3(ah, al, bih, bil, ci);
      }
      store_d(tre, tm, tn, lane, cr);
      store_d(tim, tm, tn, lane, ci);
    }
    __threadfence();
    __syncthreads();

    // ---- split pass 2: row-major hi/lo split of T (fully coalesced) ----
    for (int o = t; o < 65536; o += 256) {
      split_store(tpReHi, tpReLo, o, tre[o]);
      split_store(tpImHi, tpImLo, o, tim[o]);
    }
    __threadfence();
    __syncthreads();

    // ---- GEMM2: S = T * B^T  (matB rows give contiguous K runs) ----
    for (int tile = wave; tile < 256; tile += 8) {
      int tm = tile >> 4, tn = tile & 15;
      int hh = (lane >> 4) & 1, lan = lane & 15;
      size_t aOff = (size_t)(tm * 16 + lan) * 256 + 8 * hh;    // Tp row-major
      size_t bOff = (size_t)(tn * 16 + lan) * 256 + 8 * hh;    // matB[c'][c]
      f8 cr = {}, ci = {};
      for (int k0 = 0; k0 < 256; k0 += 32) {
        h16 arh, arl, aih, ail, bh, bl;
        load_fragA(tpReHi + aOff + k0, arh);
        load_fragA(tpReLo + aOff + k0, arl);
        load_fragA(tpImHi + aOff + k0, aih);
        load_fragA(tpImLo + aOff + k0, ail);
        load_fragB(Bh + bOff + k0, bh);
        load_fragB(Bl + bOff + k0, bl);
        cr = wmma3(arh, arl, bh, bl, cr);
        ci = wmma3(aih, ail, bh, bl, ci);
      }
      store_d(sre, tm, tn, lane, cr);
      store_d(sim, tm, tn, lane, ci);
    }
    __threadfence();
    __syncthreads();
  }

  // ---- marginals P(wire w = 0), deterministic reduction ----
  float acc[16];
#pragma unroll
  for (int w = 0; w < 16; ++w) acc[w] = 0.f;
  for (int f = t; f < 65536; f += 256) {
    float re = sre[f], im = sim[f];
    float p = re * re + im * im;
#pragma unroll
    for (int w = 0; w < 16; ++w)
      if (!((f >> (15 - w)) & 1)) acc[w] += p;
  }
#pragma unroll
  for (int w = 0; w < 16; ++w) red2[w * 256 + t] = acc[w];
  __syncthreads();
  if (t < 16) {
    float ssum = 0.f;
    for (int i = 0; i < 256; ++i) ssum += red2[t * 256 + i];
    fw[t] = ssum;
  }
  __syncthreads();

  // ---- linear head: out[s] = feats @ w_out + b_out ----
  if (t < out_dim) {
    float o = b_out[t];
#pragma unroll
    for (int w = 0; w < 16; ++w) o += fw[w] * w_out[w * out_dim + t];
    out[(size_t)s * out_dim + t] = o;
  }
}

// ---------------- Host launcher ----------------------------------------------
extern "C" void kernel_launch(void* const* d_in, const int* in_sizes, int n_in,
                              void* d_out, int out_size, void* d_ws, size_t ws_size,
                              hipStream_t stream) {
  const float* x     = (const float*)d_in[0];
  const float* wc    = (const float*)d_in[1];
  const float* wd    = (const float*)d_in[2];
  const float* w_out = (const float*)d_in[3];
  const float* b_out = (const float*)d_in[4];
  int B      = in_sizes[0] / 64;
  int cdepth = in_sizes[1] / 9;
  int ddepth = in_sizes[2] / 16;
  int out_dim = in_sizes[4];

  char* ws = (char*)d_ws;
  size_t off = 0;
  auto take = [&](size_t bytes) -> void* {
    void* p = ws + off;
    off = (off + bytes + 255) & ~(size_t)255;
    return p;
  };
  float*     cv     = (float*)take((size_t)B * 64 * sizeof(float));
  float*     pooled = (float*)take((size_t)B * 16 * sizeof(float));
  _Float16*  mats   = (_Float16*)take((size_t)ddepth * 2 * 2 * 65536 * sizeof(_Float16));
  float*     Sre    = (float*)take((size_t)B * 65536 * sizeof(float));
  float*     Sim    = (float*)take((size_t)B * 65536 * sizeof(float));
  float*     Tre    = (float*)take((size_t)B * 65536 * sizeof(float));
  float*     Tim    = (float*)take((size_t)B * 65536 * sizeof(float));
  _Float16*  Sp     = (_Float16*)take((size_t)B * 4 * 65536 * sizeof(_Float16));
  _Float16*  Tp     = (_Float16*)take((size_t)B * 4 * 65536 * sizeof(_Float16));
  (void)ws_size; (void)n_in; (void)out_size;

  qconv_kernel<<<B * 64, 256, 0, stream>>>(x, wc, cdepth, cv);
  pool_kernel<<<(B * 16 + 255) / 256, 256, 0, stream>>>(cv, pooled, B);
  build_mats_kernel<<<dim3(256, ddepth * 2), 256, 0, stream>>>(wd, mats);
  qdense_kernel<<<B, 256, 0, stream>>>(pooled, mats, Sre, Sim, Tre, Tim, Sp, Tp,
                                       w_out, b_out, (float*)d_out, ddepth, out_dim);
}
